// Efficient_CAM_Module_85658827751508
// MI455X (gfx1250) — matrix operand order repaired and checked by reading, not compiled, not run on hardware
//
#include <hip/hip_runtime.h>

// Problem constants (from reference setup_inputs)
#define B_  4
#define C_  256
#define HW_ 25600   // 160*160
#define O_  256

// GEMM tiling: 256 threads = 8 waves; block tile 128(M) x 128(N); K-step 32.
// Wave grid 4x2, each wave owns 32x64 -> 2 A frags + 4 B frags -> 8 WMMA/step.
#define BLK_M 128
#define BLK_N 128
#define BLK_K 32
#define NSPLIT_VQ 32   // split-K factor for GEMM1

typedef __attribute__((ext_vector_type(16))) __bf16         v16bf;
typedef __attribute__((ext_vector_type(8)))  float          v8f;
typedef __attribute__((ext_vector_type(16))) unsigned short us16;

#if __has_builtin(__builtin_amdgcn_global_load_async_to_lds_b128)
#define USE_ASYNC_LDS 1
typedef int aint4 __attribute__((vector_size(16)));
typedef __attribute__((address_space(1))) aint4 g_aint4;
typedef __attribute__((address_space(3))) aint4 l_aint4;
#endif

static __device__ __forceinline__ unsigned short f2bf(float f) {
  unsigned int u = __builtin_bit_cast(unsigned int, f);
  u += 0x7fffu + ((u >> 16) & 1u);            // round-to-nearest-even
  return (unsigned short)(u >> 16);
}
static __device__ __forceinline__ float bf2f(unsigned short h) {
  unsigned int u = ((unsigned int)h) << 16;
  return __builtin_bit_cast(float, u);
}

static __device__ __forceinline__ v8f wmma_bf16(v16bf a, v16bf b, v8f c) {
  // (neg_a, A, neg_b, B, c_mod, C, reuse_a, reuse_b)
  return __builtin_amdgcn_wmma_f32_16x16x32_bf16(false, a, false, b, (short)0, c,
                                                 false, false);
}

// Copy one 16-byte chunk (8 bf16) global -> LDS.
static __device__ __forceinline__ void cp_g2l_b128(unsigned short* ldst,
                                                   const unsigned short* gsrc) {
#ifdef USE_ASYNC_LDS
  __builtin_amdgcn_global_load_async_to_lds_b128((g_aint4*)gsrc, (l_aint4*)ldst,
                                                 0, 0);
#else
  *(uint4*)ldst = *(const uint4*)gsrc;
#endif
}

static __device__ __forceinline__ void wait_async_lds() {
#ifdef USE_ASYNC_LDS
#if __has_builtin(__builtin_amdgcn_s_wait_asynccnt)
  __builtin_amdgcn_s_wait_asynccnt(0);
#else
  asm volatile("s_wait_asynccnt 0x0" ::: "memory");
#endif
#endif
}

// A fragment (16x32 bf16) from LDS tile stored row-major [rows][32].
// wave32 layout: lane (g=lane/16, m=lane%16); element e -> K = (e&7) + 8g + 16*(e>>3)
static __device__ __forceinline__ v16bf frag_a(const unsigned short* At, int row0, int lane) {
  const int g = (lane >> 4) & 1;
  const int m = lane & 15;
  us16 t;
#pragma unroll
  for (int e = 0; e < 16; ++e) {
    const int k = (e & 7) + 8 * g + 16 * (e >> 3);
    t[e] = At[(row0 + m) * 32 + k];
  }
  return __builtin_bit_cast(v16bf, t);
}

// B fragment (32x16 bf16) from LDS tile stored n-major: Bt[n][k], ld 32.
// lane (g, n=lane%16); element e -> K = (e&7) + 8g + 16*(e>>3)
static __device__ __forceinline__ v16bf frag_b(const unsigned short* Bt, int n0, int lane) {
  const int g = (lane >> 4) & 1;
  const int n = n0 + (lane & 15);
  us16 t;
#pragma unroll
  for (int e = 0; e < 16; ++e) t[e] = Bt[n * 32 + (e & 7) + 8 * g + 16 * (e >> 3)];
  return __builtin_bit_cast(v16bf, t);
}

// ---------------------------------------------------------------------------
// Batched WMMA GEMM: D(M x Ncols) = A(M x K) * B(K x Ncols) [+ bias]
// BMODE 0: B source row-major [n][k] (ldb)  (contiguous -> async b128 staging)
// BMODE 1: B source row-major [k][n] (ldb)  (small transpose staging)
// BMODE 2: B built on the fly: bf16(gamma*attn[k*ldb+n] + X[k*ldb+n])
// OUTMODE 0: bf16 store ; 1: f32 store + bias[row] ; 2: f32 atomic add (split-K)
// blockIdx.z = batch * nSplit + split
// ---------------------------------------------------------------------------
template <int BMODE, int OUTMODE>
__global__ __launch_bounds__(256)
void gemm_wmma_kernel(const unsigned short* __restrict__ A, long aBatch, int lda,
                      const unsigned short* __restrict__ Bsrc, long bBatch, int ldb,
                      const float* __restrict__ X, long xBatch,
                      const float* __restrict__ gammaPtr,
                      void* __restrict__ D, long dBatch, int ldd,
                      const float* __restrict__ bias,
                      int K, int nSplit)
{
  __shared__ unsigned short At[2][BLK_M * 32];
  __shared__ unsigned short Bt[2][BLK_N * 32];

  const int z    = blockIdx.z;
  const int b    = z / nSplit;
  const int sp   = z % nSplit;
  const int kLen = K / nSplit;
  const int kBeg = sp * kLen;
  const int kEnd = kBeg + kLen;

  const int m0   = blockIdx.x * BLK_M;
  const int n0   = blockIdx.y * BLK_N;
  const int t    = threadIdx.x;
  const int lane = t & 31;
  const int wave = t >> 5;
  const int wm   = wave >> 1;   // 0..3 : 32-row band
  const int wn   = wave & 1;    // 0..1 : 64-col band

  const unsigned short* Ab = A + (long)b * aBatch;
  const unsigned short* Bb = Bsrc + (long)b * bBatch;
  const float*          Xb = (BMODE == 2) ? (X + (long)b * xBatch) : (const float*)0;
  const float gma = (BMODE == 2) ? gammaPtr[0] : 0.0f;

  v8f acc[2][4] = {};

  // ---- staging helpers ----
  auto stage = [&](int buf, int k0) {
    // A tile: BLK_M x 32 halves, row-major; 16B chunks (4 per row)
#pragma unroll
    for (int s = 0; s < (BLK_M * 4) / 256; ++s) {
      const int cid = t + 256 * s;
      const int r = cid >> 2, cs = (cid & 3) * 8;
      cp_g2l_b128(&At[buf][r * 32 + cs],
                  Ab + (long)(m0 + r) * lda + (k0 + cs));
    }
    if (BMODE == 0) {
#pragma unroll
      for (int s = 0; s < (BLK_N * 4) / 256; ++s) {
        const int cid = t + 256 * s;
        const int j = cid >> 2, cs = (cid & 3) * 8;
        cp_g2l_b128(&Bt[buf][j * 32 + cs],
                    Bb + (long)(n0 + j) * ldb + (k0 + cs));
      }
    } else if (BMODE == 1) {
#pragma unroll
      for (int s = 0; s < (BLK_N * 32) / 256; ++s) {
        const int i = t + 256 * s;
        const int j = i & (BLK_N - 1), k = i / BLK_N;
        Bt[buf][j * 32 + k] = Bb[(long)(k0 + k) * ldb + (n0 + j)];
      }
    } else {
#pragma unroll
      for (int s = 0; s < (BLK_N * 32) / 256; ++s) {
        const int i = t + 256 * s;
        const int j = i & (BLK_N - 1), k = i / BLK_N;
        const long src = (long)(k0 + k) * ldb + (n0 + j);
        Bt[buf][j * 32 + k] = f2bf(gma * bf2f(Bb[src]) + Xb[src]);
      }
    }
  };

  // ---- double-buffered main loop ----
  int p = 0;
  stage(0, kBeg);
  for (int k0 = kBeg; k0 < kEnd; k0 += BLK_K) {
    wait_async_lds();
    __syncthreads();
    if (k0 + BLK_K < kEnd) stage(1 - p, k0 + BLK_K);

    const v16bf a0 = frag_a(At[p], wm * 32 + 0,  lane);
    const v16bf a1 = frag_a(At[p], wm * 32 + 16, lane);
#pragma unroll
    for (int tj = 0; tj < 4; ++tj) {
      const v16bf bb = frag_b(Bt[p], wn * 64 + tj * 16, lane);
      acc[0][tj] = wmma_bf16(a0, bb, acc[0][tj]);
      acc[1][tj] = wmma_bf16(a1, bb, acc[1][tj]);
    }
    p ^= 1;
  }

  // ---- epilogue: C/D layout -> row = r + 8*(lane/16), col = lane%16 ----
  const int g  = (lane >> 4) & 1;
  const int cn = lane & 15;
  const long dOff = (long)b * dBatch;
#pragma unroll
  for (int ti = 0; ti < 2; ++ti) {
#pragma unroll
    for (int tj = 0; tj < 4; ++tj) {
      const v8f a = acc[ti][tj];
      const int col = n0 + wn * 64 + tj * 16 + cn;
#pragma unroll
      for (int r = 0; r < 8; ++r) {
        const int row = m0 + wm * 32 + ti * 16 + r + 8 * g;
        const long idx = dOff + (long)row * ldd + col;
        if (OUTMODE == 0) {
          ((unsigned short*)D)[idx] = f2bf(a[r]);
        } else if (OUTMODE == 1) {
          ((float*)D)[idx] = a[r] + bias[row];
        } else {
          __hip_atomic_fetch_add(((float*)D) + idx, a[r], __ATOMIC_RELAXED,
                                 __HIP_MEMORY_SCOPE_AGENT);
        }
      }
    }
  }
}

// Per-(b,c) online softmax stats over N (query softmax denominator)
__global__ __launch_bounds__(256)
void qstats_kernel(const float* __restrict__ x, float* __restrict__ mq,
                   float* __restrict__ sq)
{
  const int row = blockIdx.x;                 // b*C + c
  const float* p = x + (long)row * HW_;
  float m = -3.402823466e38f, s = 0.0f;
  for (int i = threadIdx.x; i < HW_; i += 256) {
    const float v  = p[i];
    const float mn = fmaxf(m, v);
    s = s * __expf(m - mn) + __expf(v - mn);
    m = mn;
  }
  __shared__ float sm[256], ss[256];
  sm[threadIdx.x] = m; ss[threadIdx.x] = s;
  __syncthreads();
  for (int o = 128; o > 0; o >>= 1) {
    if (threadIdx.x < (unsigned)o) {
      const float m2 = sm[threadIdx.x + o], s2 = ss[threadIdx.x + o];
      const float mn = fmaxf(sm[threadIdx.x], m2);
      ss[threadIdx.x] = ss[threadIdx.x] * __expf(sm[threadIdx.x] - mn) +
                        s2 * __expf(m2 - mn);
      sm[threadIdx.x] = mn;
    }
    __syncthreads();
  }
  if (threadIdx.x == 0) { mq[row] = sm[0]; sq[row] = ss[0]; }
}

// Build bf16 V ([b][c][n]) and bf16 Q = softmax over N ([b][d][n])
__global__ void build_v_q_kernel(const float* __restrict__ x,
                                 const float* __restrict__ mq,
                                 const float* __restrict__ sq,
                                 unsigned short* __restrict__ Vbf,
                                 unsigned short* __restrict__ Qbf, long total)
{
  for (long i = (long)blockIdx.x * blockDim.x + threadIdx.x; i < total;
       i += (long)gridDim.x * blockDim.x) {
    const long row = i / HW_;
    const float v  = x[i];
    Vbf[i] = f2bf(v);
    Qbf[i] = f2bf(__expf(v - mq[row]) / sq[row]);
  }
}

// Key softmax over C, written transposed as K^T bf16 [b][n][c]
__global__ void key_softmax_kernel(const float* __restrict__ x,
                                   unsigned short* __restrict__ Kbf)
{
  const long gid = (long)blockIdx.x * blockDim.x + threadIdx.x;
  if (gid >= (long)B_ * HW_) return;
  const int b = (int)(gid / HW_);
  const int n = (int)(gid % HW_);
  const float* p = x + (long)b * C_ * HW_ + n;
  float m = -3.402823466e38f, s = 0.0f;
  for (int c = 0; c < C_; ++c) {
    const float v  = p[(long)c * HW_];
    const float mn = fmaxf(m, v);
    s = s * __expf(m - mn) + __expf(v - mn);
    m = mn;
  }
  const float inv = 1.0f / s;
  unsigned short* o = Kbf + ((long)b * HW_ + n) * C_;
  for (int c = 0; c < C_; ++c)
    o[c] = f2bf(__expf(p[(long)c * HW_] - m) * inv);
}

__global__ void convert_bf16_kernel(const float* __restrict__ src,
                                    unsigned short* __restrict__ dst, int n)
{
  const int i = blockIdx.x * blockDim.x + threadIdx.x;
  if (i < n) dst[i] = f2bf(src[i]);
}

__global__ void zero_f32_kernel(float* __restrict__ p, int n)
{
  const int i = blockIdx.x * blockDim.x + threadIdx.x;
  if (i < n) p[i] = 0.0f;
}

// ---------------------------------------------------------------------------
extern "C" void kernel_launch(void* const* d_in, const int* in_sizes, int n_in,
                              void* d_out, int out_size, void* d_ws, size_t ws_size,
                              hipStream_t stream)
{
  (void)in_sizes; (void)n_in; (void)out_size; (void)ws_size;
  const float* x      = (const float*)d_in[0];   // (B, C, H, W)
  const float* conv_w = (const float*)d_in[1];   // (O, C)
  const float* conv_b = (const float*)d_in[2];   // (O,)
  const float* gamma  = (const float*)d_in[3];   // (1,)
  float* y = (float*)d_out;                      // (B, O, H, W) f32

  // Workspace layout (bytes, 256-aligned)
  char* ws = (char*)d_ws;
  float*          mq    = (float*)(ws + 0);                  //   4 KB
  float*          sq    = (float*)(ws + 4096);               //   4 KB
  unsigned short* wbf   = (unsigned short*)(ws + 8192);      // 128 KB
  float*          vqf32 = (float*)(ws + 139264);             //   1 MB : (B,C,C) f32
  unsigned short* vqbf  = (unsigned short*)(ws + 1187840);   // 512 KB : (B,C,C) bf16
  unsigned short* Vbf   = (unsigned short*)(ws + 1712128);   //  50 MB : (B,C,N)
  unsigned short* Kbf   = (unsigned short*)(ws + 54140928);  //  50 MB : (B,N,C)
  unsigned short* Qbf   = (unsigned short*)(ws + 106569728); //  50 MB : (B,C,N)
  unsigned short* attn  = Qbf;  // GEMM2 output aliases Q (Q dead after GEMM1)

  const long CN = (long)C_ * HW_;   // per-batch V/Q/x elems
  const long NC = (long)HW_ * C_;   // per-batch K^T/attn elems
  const int  VQN = B_ * C_ * C_;

  // 0) conv_w -> bf16 ; zero split-K accumulator
  convert_bf16_kernel<<<(O_ * C_ + 255) / 256, 256, 0, stream>>>(conv_w, wbf, O_ * C_);
  zero_f32_kernel<<<(VQN + 255) / 256, 256, 0, stream>>>(vqf32, VQN);
  // 1) query-softmax stats (max/sumexp over N per (b,c))
  qstats_kernel<<<B_ * C_, 256, 0, stream>>>(x, mq, sq);
  // 2) bf16 V and Q
  build_v_q_kernel<<<4096, 256, 0, stream>>>(x, mq, sq, Vbf, Qbf, (long)B_ * CN);
  // 3) key softmax -> K^T bf16 [b][n][c]
  key_softmax_kernel<<<(B_ * HW_ + 255) / 256, 256, 0, stream>>>(x, Kbf);

  // 4) GEMM1 (split-K x32): vq(CxC) = V(CxN) * Q^T ; f32 atomic accumulate
  {
    dim3 grid(C_ / BLK_M, C_ / BLK_N, B_ * NSPLIT_VQ);
    gemm_wmma_kernel<0, 2><<<grid, 256, 0, stream>>>(
        Vbf, CN, HW_, Qbf, CN, HW_, (const float*)0, 0, gamma,
        vqf32, (long)C_ * C_, C_, (const float*)0, HW_, NSPLIT_VQ);
  }
  // 4b) vq f32 -> bf16
  convert_bf16_kernel<<<(VQN + 255) / 256, 256, 0, stream>>>(vqf32, vqbf, VQN);

  // 5) GEMM2: attn(NxC) = K^T(NxC) * vq(CxC)   (B src vqbf [c][d] -> BMODE 1)
  {
    dim3 grid(HW_ / BLK_M, C_ / BLK_N, B_);
    gemm_wmma_kernel<1, 0><<<grid, 256, 0, stream>>>(
        Kbf, NC, C_, vqbf, (long)C_ * C_, C_, (const float*)0, 0, gamma,
        attn, NC, C_, (const float*)0, C_, 1);
  }
  // 6) GEMM3: y(OxN) = W(OxC) * res(CxN),  res = gamma*attn_reinterp + x (BMODE 2)
  //    attn (B,N,C) reinterpreted as (B,C,N): element [c][n] = attn_flat[c*N + n]
  {
    dim3 grid(O_ / BLK_M, HW_ / BLK_N, B_);
    gemm_wmma_kernel<2, 1><<<grid, 256, 0, stream>>>(
        wbf, 0, C_, attn, NC, HW_, x, CN, gamma,
        y, (long)O_ * HW_, HW_, conv_b, C_, 1);
  }
}
